// FCOSModule_25795573580544
// MI455X (gfx1250) — matrix operand order repaired and checked by reading, not compiled, not run on hardware
//
#include <hip/hip_runtime.h>
#include <hip/hip_bf16.h>

typedef _Float16 v16h __attribute__((ext_vector_type(16)));
typedef _Float16 v8h  __attribute__((ext_vector_type(8)));
typedef float    v8f  __attribute__((ext_vector_type(8)));

#define NB   32     // batch
#define CCH  128    // channels
#define NCLSX 20
#define TOPKX 100

// ---------------------------------------------------------------------------
// Tiled transpose + convert: in [C][T] f32  ->  out [T][C] f16 (per image)
// grid (T/32, C/32, N), block (32, 8). Coalesced on both sides.
// ---------------------------------------------------------------------------
__global__ void transpose_f32_to_f16(const float* __restrict__ in,
                                     _Float16* __restrict__ out,
                                     int C, int T) {
    __shared__ float tile[32][33];
    const int t0 = blockIdx.x * 32;
    const int c0 = blockIdx.y * 32;
    const int img = blockIdx.z;
    const int tx = threadIdx.x, ty = threadIdx.y;
    const float* ip = in + (size_t)img * C * T;
#pragma unroll
    for (int j = 0; j < 4; ++j) {
        int c = c0 + ty + j * 8;
        tile[ty + j * 8][tx] = ip[(size_t)c * T + t0 + tx];
    }
    __syncthreads();
    _Float16* op = out + (size_t)img * T * C;
#pragma unroll
    for (int j = 0; j < 4; ++j) {
        int t = t0 + ty + j * 8;
        op[(size_t)t * C + c0 + tx] = (_Float16)tile[tx][ty + j * 8];
    }
}

// ---------------------------------------------------------------------------
// Fold BN (eval mode) + conv bias into per-channel scale/shift
// ---------------------------------------------------------------------------
__global__ void bn_prep_kernel(const float* __restrict__ g,
                               const float* __restrict__ beta,
                               const float* __restrict__ m,
                               const float* __restrict__ v,
                               const float* __restrict__ b,
                               float* __restrict__ scale,
                               float* __restrict__ shift, int C) {
    int i = blockIdx.x * blockDim.x + threadIdx.x;
    if (i >= C) return;
    float s = g[i] * rsqrtf(v[i] + 1e-5f);
    scale[i] = s;
    shift[i] = (b[i] - m[i]) * s + beta[i];
}

// ---------------------------------------------------------------------------
// Pack conv weights [Cout][Cin][ktaps] (f32) into WMMA A-fragment order (f16).
// out[((tap*ntiles + tile)*nkblk + kb)*32 + lane][e], e = 0..15
// A 16x16x32 f16 fragment, contiguous k order (not ISA table order): g=lane>>4, M = tile*16+(lane&15)
//   K = e + 16*g + kb*32   (K == cin per tap)
// ---------------------------------------------------------------------------
__global__ void pack_a_kernel(const float* __restrict__ W,
                              _Float16* __restrict__ out,
                              int Cout, int Cin, int ktaps,
                              int ntiles, int nkblk) {
    int idx = blockIdx.x * blockDim.x + threadIdx.x;
    int total = ktaps * ntiles * nkblk * 32 * 16;
    if (idx >= total) return;
    int e    = idx & 15;
    int lane = (idx >> 4) & 31;
    int kb   = (idx >> 9) % nkblk;
    int tile = (idx / (512 * nkblk)) % ntiles;
    int tap  =  idx / (512 * nkblk * ntiles);
    int g = lane >> 4;
    int m = tile * 16 + (lane & 15);
    int r = e >> 1, s = e & 1;
    int koff = (r < 4) ? (2 * r + s) : (16 + 2 * (r - 4) + s);
    int K = kb * 32 + e + 16 * g; (void)koff;
    float w = 0.f;
    if (m < Cout && K < Cin) w = W[(m * Cin + K) * ktaps + tap];
    out[idx] = (_Float16)w;
}

// ---------------------------------------------------------------------------
// Conv1d-as-GEMM, NHWC ([T][C]) activations, no LDS:
//  - B fragment = 16 contiguous f16 (K = cin) per lane, 32B-aligned direct load
//  - one wave computes TWO 16x16 t-tiles sharing A fragments (24 WMMAs for k3)
//  - fully unrolled via template params
// postop: 0 = +shift, f32 out | 1 = BN+ReLU, f16 out | 2 = exp(es*(y+shift)), f32
// ---------------------------------------------------------------------------
template <int KTAPS, int NKBLK>
__global__ __launch_bounds__(32)
void conv_wmma_kernel(const _Float16* __restrict__ xin, int T,
                      int rowStrideIn, int colOffIn, size_t strideImgIn,
                      const _Float16* __restrict__ packA,
                      int ntiles, int CoutValid,
                      const float* __restrict__ scale,
                      const float* __restrict__ shift,
                      int postop, const float* __restrict__ expScalePtr,
                      _Float16* __restrict__ outH, int rowStrideOutH,
                      int colOffOut, size_t strideImgOutH,
                      float* __restrict__ outF, int rowStrideOutF,
                      size_t strideImgOutF) {
    const int t0   = blockIdx.x * 32;
    const int tile = blockIdx.y;
    const int img  = blockIdx.z;
    const int lane = threadIdx.x;
    const int g    = lane >> 4;
    const int nlo  = lane & 15;
    constexpr int PAD = KTAPS / 2;

    const _Float16* x = xin + (size_t)img * strideImgIn + colOffIn;
    const v16h* pA = (const v16h*)packA;

    v8f acc0 = {};
    v8f acc1 = {};
#pragma unroll
    for (int tap = 0; tap < KTAPS; ++tap) {
        const int tp0 = t0 + nlo + tap - PAD;
        const int tp1 = tp0 + 16;
#pragma unroll
        for (int kb = 0; kb < NKBLK; ++kb) {
            v16h a = pA[(((size_t)tap * ntiles + tile) * NKBLK + kb) * 32 + lane];
            const int co = kb * 32 + 16 * g;
            v16h b0 = {};
            v16h b1 = {};
            if ((unsigned)tp0 < (unsigned)T)
                b0 = *(const v16h*)(x + (size_t)tp0 * rowStrideIn + co);
            if ((unsigned)tp1 < (unsigned)T)
                b1 = *(const v16h*)(x + (size_t)tp1 * rowStrideIn + co);
            acc0 = __builtin_amdgcn_wmma_f32_16x16x32_f16(
                false, a, false, b0, (short)0, acc0, false, false);
            acc1 = __builtin_amdgcn_wmma_f32_16x16x32_f16(
                false, a, false, b1, (short)0, acc1, false, false);
        }
    }

    // epilogue: lane holds channels cbase..cbase+7 at column t
    const int cbase = tile * 16 + 8 * g;
    const float es = (postop == 2) ? expScalePtr[0] : 0.f;
#pragma unroll
    for (int u = 0; u < 2; ++u) {
        const v8f acc = u ? acc1 : acc0;
        const int t = t0 + u * 16 + nlo;
        if (postop == 1) {
            v8f sc = *(const v8f*)&scale[cbase];
            v8f sh = *(const v8f*)&shift[cbase];
            v8h hv;
#pragma unroll
            for (int r = 0; r < 8; ++r)
                hv[r] = (_Float16)fmaxf(acc[r] * sc[r] + sh[r], 0.f);
            *(v8h*)(outH + (size_t)img * strideImgOutH +
                    (size_t)t * rowStrideOutH + colOffOut + cbase) = hv;
        } else {
#pragma unroll
            for (int r = 0; r < 8; ++r) {
                int ch = cbase + r;
                if (ch >= CoutValid) continue;
                float y = acc[r] + (shift ? shift[ch] : 0.f);
                if (postop == 2) y = expf(es * y);
                outF[(size_t)img * strideImgOutF + (size_t)t * rowStrideOutF + ch] = y;
            }
        }
    }
}

// ---------------------------------------------------------------------------
// iou2: 64 -> 1, k=1. iouh is [T][64] f16 -> contiguous dot per position.
// ---------------------------------------------------------------------------
__global__ void iou2_kernel(const _Float16* __restrict__ iouh, int T,
                            const float* __restrict__ w,
                            const float* __restrict__ b,
                            float* __restrict__ outF, int total) {
    int idx = blockIdx.x * blockDim.x + threadIdx.x;
    if (idx >= total) return;
    const _Float16* h = iouh + (size_t)idx * 64;
    float s = 0.f;
#pragma unroll
    for (int c = 0; c < 64; ++c) s += (float)h[c] * w[c];
    outF[idx] = s + b[0];
}

// ---------------------------------------------------------------------------
// flat[img][t*NCLS + c] = sigmoid(logit) > thr ? sig*sig(iou) : -1
// logitsF is [T][NCLS] so this is fully coalesced.
// ---------------------------------------------------------------------------
__global__ void score_flat_kernel(const float* __restrict__ logitsF,
                                  const float* __restrict__ iouF, int T,
                                  float* __restrict__ flat, int total) {
    int idx = blockIdx.x * blockDim.x + threadIdx.x;
    if (idx >= total) return;
    float iou_s = 1.f / (1.f + expf(-iouF[idx]));
    const float* lg = logitsF + (size_t)idx * NCLSX;
    float* fo = flat + (size_t)idx * NCLSX;
#pragma unroll
    for (int c = 0; c < NCLSX; ++c) {
        float cs = 1.f / (1.f + expf(-lg[c]));
        fo[c] = (cs > 0.05f) ? cs * iou_s : -1.0f;
    }
}

// ---------------------------------------------------------------------------
// Per-image top-K (K=100), jax tie-break (lowest index), detection decode.
// One block per image. Destroys flat. reg is [T][2].
// ---------------------------------------------------------------------------
__global__ void topk_kernel(float* __restrict__ flat, int T,
                            const float* __restrict__ reg,
                            const float* __restrict__ loc,
                            float* __restrict__ dets,
                            float* __restrict__ scores,
                            float* __restrict__ labels,
                            float* __restrict__ valids, int level) {
    __shared__ float sv[256];
    __shared__ int   si[256];
    const int img = blockIdx.x;
    const int tid = threadIdx.x;
    float* f = flat + (size_t)img * T * NCLSX;
    const int len = T * NCLSX;
    const float* regb = reg + (size_t)img * 2 * T;

    for (int k = 0; k < TOPKX; ++k) {
        float bv = -3.4e38f;
        int   bi = len;
        for (int i = tid; i < len; i += 256) {
            float v = f[i];
            if (v > bv || (v == bv && i < bi)) { bv = v; bi = i; }
        }
        sv[tid] = bv; si[tid] = bi;
        __syncthreads();
        for (int s = 128; s > 0; s >>= 1) {
            if (tid < s) {
                if (sv[tid + s] > sv[tid] ||
                    (sv[tid + s] == sv[tid] && si[tid + s] < si[tid])) {
                    sv[tid] = sv[tid + s]; si[tid] = si[tid + s];
                }
            }
            __syncthreads();
        }
        if (tid == 0) {
            float val = sv[0];
            int   idx = si[0];
            if (idx >= len) idx = 0;
            f[idx] = -3.0e38f;  // consume
            int li  = idx / NCLSX;
            int lab = idx - li * NCLSX + 1;
            float r0 = regb[li * 2 + 0];
            float r1 = regb[li * 2 + 1];
            float gl = loc[li];
            float d0 = fminf(fmaxf((gl - r0) * (1.f / 32.f), 0.f), 1.f);
            float d1 = fminf(fmaxf((gl + r1) * (1.f / 32.f), 0.f), 1.f);
            bool vld = (val > 0.f) && (d1 - d0 >= 0.f);
            int row = img * (3 * TOPKX) + level * TOPKX + k;
            dets[row * 2 + 0] = d0;
            dets[row * 2 + 1] = d1;
            scores[row] = vld ? sqrtf(fmaxf(val, 1e-12f)) : 0.f;
            labels[row] = (float)lab;
            valids[row] = vld ? 1.f : 0.f;
        }
        __syncthreads();
    }
}

// ---------------------------------------------------------------------------
extern "C" void kernel_launch(void* const* d_in, const int* in_sizes, int n_in,
                              void* d_out, int out_size, void* d_ws, size_t ws_size,
                              hipStream_t stream) {
    (void)in_sizes; (void)n_in; (void)out_size; (void)ws_size;
    const int Ts[3] = {4096, 2048, 1024};
    const int Tmax = 4096;

    const float* feat[3] = {(const float*)d_in[0], (const float*)d_in[2], (const float*)d_in[4]};
    const float* loc[3]  = {(const float*)d_in[1], (const float*)d_in[3], (const float*)d_in[5]};
    const float* cls_w = (const float*)d_in[6];
    const float* cls_b = (const float*)d_in[7];
    const float* cls_g = (const float*)d_in[8];
    const float* cls_beta = (const float*)d_in[9];
    const float* cls_m = (const float*)d_in[10];
    const float* cls_v = (const float*)d_in[11];
    const float* box_w = (const float*)d_in[12];
    const float* box_b = (const float*)d_in[13];
    const float* box_g = (const float*)d_in[14];
    const float* box_beta = (const float*)d_in[15];
    const float* box_m = (const float*)d_in[16];
    const float* box_v = (const float*)d_in[17];
    const float* logits_w = (const float*)d_in[18];
    const float* logits_b = (const float*)d_in[19];
    const float* bbox_w = (const float*)d_in[20];
    const float* bbox_b = (const float*)d_in[21];
    const float* mix_w = (const float*)d_in[22];
    const float* mix_b = (const float*)d_in[23];
    const float* mix_g = (const float*)d_in[24];
    const float* mix_beta = (const float*)d_in[25];
    const float* mix_m = (const float*)d_in[26];
    const float* mix_v = (const float*)d_in[27];
    const float* iou1_w = (const float*)d_in[28];
    const float* iou1_b = (const float*)d_in[29];
    const float* iou1_g = (const float*)d_in[30];
    const float* iou1_beta = (const float*)d_in[31];
    const float* iou1_m = (const float*)d_in[32];
    const float* iou1_v = (const float*)d_in[33];
    const float* iou2_w = (const float*)d_in[34];
    const float* iou2_b = (const float*)d_in[35];
    const float* scalesP = (const float*)d_in[36];

    // ---- workspace carve (256B aligned) ----
    char* base = (char*)d_ws;
    size_t off = 0;
    auto carve = [&](size_t bytes) -> char* {
        char* p = base + off;
        off = (off + bytes + 255) & ~(size_t)255;
        return p;
    };
    _Float16* xh    = (_Float16*)carve((size_t)NB * Tmax * CCH * 2);      // [N][T][128]
    _Float16* act1  = (_Float16*)carve((size_t)NB * Tmax * CCH * 2);      // [N][T][128]
    _Float16* ctbt  = (_Float16*)carve((size_t)NB * Tmax * 2 * CCH * 2);  // [N][T][256]
    _Float16* iouh  = (_Float16*)carve((size_t)NB * Tmax * 64 * 2);       // [N][T][64]
    float* logitsF  = (float*)carve((size_t)NB * Tmax * NCLSX * 4);       // [N][T][20]
    float* regF     = (float*)carve((size_t)NB * Tmax * 2 * 4);           // [N][T][2]
    float* iouF     = (float*)carve((size_t)NB * Tmax * 4);               // [N][T]
    float* flat     = (float*)carve((size_t)NB * Tmax * NCLSX * 4);
    _Float16* pCls0 = (_Float16*)carve(3 * 8 * 4 * 512 * 2);
    _Float16* pCls1 = (_Float16*)carve(3 * 8 * 4 * 512 * 2);
    _Float16* pBox0 = (_Float16*)carve(3 * 8 * 4 * 512 * 2);
    _Float16* pBox1 = (_Float16*)carve(3 * 8 * 4 * 512 * 2);
    _Float16* pLog  = (_Float16*)carve(3 * 2 * 4 * 512 * 2);
    _Float16* pBbx  = (_Float16*)carve(3 * 1 * 4 * 512 * 2);
    _Float16* pMix  = (_Float16*)carve(1 * 8 * 8 * 512 * 2);
    _Float16* pIou1 = (_Float16*)carve(3 * 4 * 4 * 512 * 2);
    float* scCls0 = (float*)carve(CCH * 4);  float* shCls0 = (float*)carve(CCH * 4);
    float* scCls1 = (float*)carve(CCH * 4);  float* shCls1 = (float*)carve(CCH * 4);
    float* scBox0 = (float*)carve(CCH * 4);  float* shBox0 = (float*)carve(CCH * 4);
    float* scBox1 = (float*)carve(CCH * 4);  float* shBox1 = (float*)carve(CCH * 4);
    float* scMix  = (float*)carve(CCH * 4);  float* shMix  = (float*)carve(CCH * 4);
    float* scIou1 = (float*)carve(64 * 4);   float* shIou1 = (float*)carve(64 * 4);

    float* out    = (float*)d_out;
    float* dets   = out;                        // [N][300][2]
    float* scores = out + (size_t)NB * 300 * 2; // [N][300]
    float* labels = scores + (size_t)NB * 300;  // [N][300]
    float* valids = labels + (size_t)NB * 300;  // [N][300]

    // ---- pack weights into WMMA A-fragment layout ----
    auto packLaunch = [&](const float* W, _Float16* dst, int Cout, int Cin,
                          int ktaps, int ntiles, int nkblk) {
        int total = ktaps * ntiles * nkblk * 512;
        pack_a_kernel<<<(total + 255) / 256, 256, 0, stream>>>(
            W, dst, Cout, Cin, ktaps, ntiles, nkblk);
    };
    packLaunch(cls_w,                 pCls0, 128, 128, 3, 8, 4);
    packLaunch(cls_w + 128 * 128 * 3, pCls1, 128, 128, 3, 8, 4);
    packLaunch(box_w,                 pBox0, 128, 128, 3, 8, 4);
    packLaunch(box_w + 128 * 128 * 3, pBox1, 128, 128, 3, 8, 4);
    packLaunch(logits_w,              pLog,   20, 128, 3, 2, 4);
    packLaunch(bbox_w,                pBbx,    2, 128, 3, 1, 4);
    packLaunch(mix_w,                 pMix,  128, 256, 1, 8, 8);
    packLaunch(iou1_w,                pIou1,  64, 128, 3, 4, 4);

    // ---- fold BN params ----
    bn_prep_kernel<<<1, 128, 0, stream>>>(cls_g, cls_beta, cls_m, cls_v, cls_b, scCls0, shCls0, 128);
    bn_prep_kernel<<<1, 128, 0, stream>>>(cls_g + 128, cls_beta + 128, cls_m + 128, cls_v + 128, cls_b + 128, scCls1, shCls1, 128);
    bn_prep_kernel<<<1, 128, 0, stream>>>(box_g, box_beta, box_m, box_v, box_b, scBox0, shBox0, 128);
    bn_prep_kernel<<<1, 128, 0, stream>>>(box_g + 128, box_beta + 128, box_m + 128, box_v + 128, box_b + 128, scBox1, shBox1, 128);
    bn_prep_kernel<<<1, 128, 0, stream>>>(mix_g, mix_beta, mix_m, mix_v, mix_b, scMix, shMix, 128);
    bn_prep_kernel<<<1, 64, 0, stream>>>(iou1_g, iou1_beta, iou1_m, iou1_v, iou1_b, scIou1, shIou1, 64);

    for (int l = 0; l < 3; ++l) {
        const int T = Ts[l];
        const int nT2 = T / 32;  // two 16-col tiles per wave
        const size_t sX  = (size_t)T * CCH;     // [T][128] halves
        const size_t sCB = (size_t)T * 2 * CCH; // [T][256] halves

        transpose_f32_to_f16<<<dim3(T / 32, CCH / 32, NB), dim3(32, 8), 0, stream>>>(
            feat[l], xh, CCH, T);

        // cls tower
        conv_wmma_kernel<3, 4><<<dim3(nT2, 8, NB), 32, 0, stream>>>(
            xh, T, CCH, 0, sX, pCls0, 8, 128, scCls0, shCls0, 1, nullptr,
            act1, CCH, 0, sX, nullptr, 0, 0);
        conv_wmma_kernel<3, 4><<<dim3(nT2, 8, NB), 32, 0, stream>>>(
            act1, T, CCH, 0, sX, pCls1, 8, 128, scCls1, shCls1, 1, nullptr,
            ctbt, 2 * CCH, 0, sCB, nullptr, 0, 0);
        // box tower
        conv_wmma_kernel<3, 4><<<dim3(nT2, 8, NB), 32, 0, stream>>>(
            xh, T, CCH, 0, sX, pBox0, 8, 128, scBox0, shBox0, 1, nullptr,
            act1, CCH, 0, sX, nullptr, 0, 0);
        conv_wmma_kernel<3, 4><<<dim3(nT2, 8, NB), 32, 0, stream>>>(
            act1, T, CCH, 0, sX, pBox1, 8, 128, scBox1, shBox1, 1, nullptr,
            ctbt, 2 * CCH, CCH, sCB, nullptr, 0, 0);
        // logits head (ct columns 0..127 of ctbt)
        conv_wmma_kernel<3, 4><<<dim3(nT2, 2, NB), 32, 0, stream>>>(
            ctbt, T, 2 * CCH, 0, sCB, pLog, 2, NCLSX, nullptr, logits_b, 0, nullptr,
            nullptr, 0, 0, 0, logitsF, NCLSX, (size_t)T * NCLSX);
        // bbox head (bt columns 128..255), reg = exp(scale_l * (conv + b))
        conv_wmma_kernel<3, 4><<<dim3(nT2, 1, NB), 32, 0, stream>>>(
            ctbt, T, 2 * CCH, CCH, sCB, pBbx, 1, 2, nullptr, bbox_b, 2, scalesP + l,
            nullptr, 0, 0, 0, regF, 2, (size_t)T * 2);
        // mix: 1x1 conv over concat(ct,bt) = 256 channels
        conv_wmma_kernel<1, 8><<<dim3(nT2, 8, NB), 32, 0, stream>>>(
            ctbt, T, 2 * CCH, 0, sCB, pMix, 8, 128, scMix, shMix, 1, nullptr,
            act1, CCH, 0, sX, nullptr, 0, 0);
        // iou1: 128 -> 64, k=3
        conv_wmma_kernel<3, 4><<<dim3(nT2, 4, NB), 32, 0, stream>>>(
            act1, T, CCH, 0, sX, pIou1, 4, 64, scIou1, shIou1, 1, nullptr,
            iouh, 64, 0, (size_t)T * 64, nullptr, 0, 0);
        // iou2: 64 -> 1 scalar dot (contiguous per position)
        int npos = NB * T;
        iou2_kernel<<<(npos + 255) / 256, 256, 0, stream>>>(
            iouh, T, iou2_w, iou2_b, iouF, npos);
        // postprocess
        score_flat_kernel<<<(npos + 255) / 256, 256, 0, stream>>>(
            logitsF, iouF, T, flat, npos);
        topk_kernel<<<NB, 256, 0, stream>>>(
            flat, T, regF, loc[l], dets, scores, labels, valids, l);
    }
}
